// SparseLinear_65841848648012
// MI455X (gfx1250) — matrix operand order repaired and checked by reading, not compiled, not run on hardware
//
#include <hip/hip_runtime.h>
#include <hip/hip_bf16.h>

typedef __attribute__((ext_vector_type(4)))  float        v4f;
typedef __attribute__((ext_vector_type(8)))  float        v8f;
typedef __attribute__((ext_vector_type(4)))  unsigned int v4u;
typedef __attribute__((ext_vector_type(16))) __bf16       v16bf;
typedef int v4i_b __attribute__((vector_size(4 * sizeof(int))));  // async builtin param type

#define BS       32
#define OB       128          // OUT_FEATURES / BS
#define KDIM     4096         // IN_FEATURES
#define NDIM     4096         // OUT_FEATURES
#define MTILE    128
#define NTHREADS 256
#define PITCH    40           // bf16/row: 80B = mult of 16B (async b128) ; 20-bank stride

#define LDS_AS    __attribute__((address_space(3)))
#define GLOBAL_AS __attribute__((address_space(1)))

#define ASYNC_B128(gsrc, ldst)                                                 \
    __builtin_amdgcn_global_load_async_to_lds_b128(                            \
        (GLOBAL_AS v4i_b*)(gsrc), (LDS_AS v4i_b*)(ldst), 0, 0)

#if defined(__has_builtin)
#  if __has_builtin(__builtin_amdgcn_s_wait_asynccnt)
#    define WAIT_ASYNC0() __builtin_amdgcn_s_wait_asynccnt(0)
#  endif
#endif
#ifndef WAIT_ASYNC0
#  define WAIT_ASYNC0() asm volatile("s_wait_asynccnt 0" ::: "memory")
#endif

// ---------------- bf16 split helpers (RNE) ----------------
static __device__ __forceinline__ unsigned short f32_to_bf16_rne(float f) {
    unsigned u = __float_as_uint(f);
    u += 0x7fffu + ((u >> 16) & 1u);
    return (unsigned short)(u >> 16);
}
static __device__ __forceinline__ float bf16_to_f32(unsigned short h) {
    return __uint_as_float(((unsigned)h) << 16);
}

// ---------------------------------------------------------------------------
// Kernel 0a: split x into bf16 hi/lo planes (one-time, memory-bound pass).
// ---------------------------------------------------------------------------
__global__ __launch_bounds__(256)
void SparseLinear_split_x(const float* __restrict__ x,
                          unsigned short* __restrict__ hi,
                          unsigned short* __restrict__ lo, long long n4) {
    long long i = (long long)blockIdx.x * 256 + threadIdx.x;
    if (i >= n4) return;
    v4f v = ((const v4f*)x)[i];
    float f[4] = {v.x, v.y, v.z, v.w};
    union { unsigned short s[4]; unsigned long long u; } ph, pl;
    #pragma unroll
    for (int j = 0; j < 4; ++j) {
        unsigned short h = f32_to_bf16_rne(f[j]);
        ph.s[j] = h;
        pl.s[j] = f32_to_bf16_rne(f[j] - bf16_to_f32(h));
    }
    *(unsigned long long*)(hi + 4 * i) = ph.u;
    *(unsigned long long*)(lo + 4 * i) = pl.u;
}

// ---------------------------------------------------------------------------
// Kernel 0b: split + transpose w_blocks: wT[n][j][i] = split(w[n][i][j]).
// Transposed [N][K] layout makes B fragments contiguous 16B LDS loads.
// ---------------------------------------------------------------------------
__global__ __launch_bounds__(256)
void SparseLinear_split_wT(const float* __restrict__ w,
                           unsigned short* __restrict__ hiT,
                           unsigned short* __restrict__ loT) {
    const int n = blockIdx.x;
    const int t = threadIdx.x;
    const int i  = t >> 3;            // K row 0..31
    const int j0 = (t & 7) * 4;       // N col group
    const float* src = w + (size_t)n * (BS * BS) + i * BS + j0;
    unsigned short* hb = hiT + (size_t)n * (BS * BS);
    unsigned short* lb = loT + (size_t)n * (BS * BS);
    v4f v = *(const v4f*)src;
    float f[4] = {v.x, v.y, v.z, v.w};
    #pragma unroll
    for (int jj = 0; jj < 4; ++jj) {
        unsigned short h = f32_to_bf16_rne(f[jj]);
        hb[(j0 + jj) * BS + i] = h;
        lb[(j0 + jj) * BS + i] = f32_to_bf16_rne(f[jj] - bf16_to_f32(h));
    }
}

// ---------------------------------------------------------------------------
// Kernel A: deterministic CSC build over output block-columns.
// ---------------------------------------------------------------------------
__global__ __launch_bounds__(128)
void SparseLinear_build_csc(const int* __restrict__ ks, int nnz,
                            int* __restrict__ col_ptr, int* __restrict__ col_idx) {
    const int t = threadIdx.x;
    __shared__ int counts[OB];
    __shared__ int offs[OB + 1];
    int cnt = 0;
    for (int n = 0; n < nnz; ++n) cnt += (ks[n] == t) ? 1 : 0;
    counts[t] = cnt;
    __syncthreads();
    if (t == 0) {
        int run = 0;
        for (int i = 0; i < OB; ++i) { offs[i] = run; run += counts[i]; }
        offs[OB] = run;
    }
    __syncthreads();
    int pos = offs[t];
    for (int n = 0; n < nnz; ++n)
        if (ks[n] == t) col_idx[pos++] = n;
    col_ptr[t] = offs[t];
    if (t == 0) col_ptr[OB] = offs[OB];
}

// ---------------------------------------------------------------------------
// Async-stage one sparse block (hi/lo x tile + hi/lo wT block) into LDS.
// 5 GLOBAL_LOAD_ASYNC_TO_LDS_B128 per thread, tracked by ASYNCcnt.
// ---------------------------------------------------------------------------
static __device__ __forceinline__ void request_block(
    const unsigned short* __restrict__ xhi, const unsigned short* __restrict__ xlo,
    const unsigned short* __restrict__ whT, const unsigned short* __restrict__ wlT,
    int c, int n, size_t mBase, int tid,
    unsigned short (*__restrict__ xh)[PITCH], unsigned short (*__restrict__ xl)[PITCH],
    unsigned short (*__restrict__ wh)[PITCH], unsigned short (*__restrict__ wl)[PITCH]) {
    const int xrow = tid >> 2;               // 0..63
    const int xoff = (tid & 3) * 8;          // bf16 col: 0,8,16,24 (16B chunks)
    #pragma unroll
    for (int i = 0; i < 2; ++i) {
        const int r = xrow + 64 * i;
        const size_t g = (mBase + r) * KDIM + (size_t)c * BS + xoff;
        ASYNC_B128(xhi + g, &xh[r][xoff]);
        ASYNC_B128(xlo + g, &xl[r][xoff]);
    }
    // wT planes: threads 0..127 -> hi, 128..255 -> lo (no EXEC divergence).
    const int wrow = (tid & 127) >> 2;       // 0..31
    const int woff = (tid & 3) * 8;
    const unsigned short* wsrc = (tid < 128) ? whT : wlT;
    unsigned short (*wdst)[PITCH] = (tid < 128) ? wh : wl;
    ASYNC_B128(wsrc + (size_t)n * (BS * BS) + wrow * BS + woff, &wdst[wrow][woff]);
}

// ---------------------------------------------------------------------------
// Kernel B: block-sparse GEMM, bf16x3 split (hi*hi + lo*hi + hi*lo, f32 acc),
// double-buffered async LDS. grid = (128 columns, M/128 tiles), 8 waves.
// Per block per wave: 12 ds_load_b128 + 6 v_wmma_f32_16x16x32_bf16.
// ---------------------------------------------------------------------------
__global__ __launch_bounds__(NTHREADS)
void SparseLinear_bsmm(const unsigned short* __restrict__ xhi,
                       const unsigned short* __restrict__ xlo,
                       const unsigned short* __restrict__ whT,
                       const unsigned short* __restrict__ wlT,
                       const int* __restrict__ block_cs,
                       const int* __restrict__ col_ptr,
                       const int* __restrict__ col_idx,
                       float* __restrict__ out) {
    __shared__ __align__(16) unsigned short xhl[2][2][MTILE][PITCH]; // 40 KB
    __shared__ __align__(16) unsigned short whl[2][2][BS][PITCH];    // 10 KB

    const int kblk = blockIdx.x;
    const int mt   = blockIdx.y;
    const int tid  = threadIdx.x;
    const int lane = tid & 31;
    const int wave = tid >> 5;
    const int m0   = wave * 16;
    const int half = lane >> 4;            // 0 | 1
    const int l15  = lane & 15;

    const int p0 = col_ptr[kblk];
    const int p1 = col_ptr[kblk + 1];

    v8f acc0 = {};                          // N columns [0..15]
    v8f acc1 = {};                          // N columns [16..31]

    const size_t mBase = (size_t)mt * MTILE;

    if (p0 < p1) {
        const int n = col_idx[p0];
        request_block(xhi, xlo, whT, wlT, block_cs[n], n, mBase, tid,
                      xhl[0][0], xhl[0][1], whl[0][0], whl[0][1]);
    }

    union Frag { v4u q[2]; v16bf v; };

    for (int p = p0; p < p1; ++p) {
        const int buf = (p - p0) & 1;
        WAIT_ASYNC0();                      // my copies into `buf` done
        __syncthreads();                    // all waves' copies visible; buf^1 free

        if (p + 1 < p1) {
            const int n1 = col_idx[p + 1];
            request_block(xhi, xlo, whT, wlT, block_cs[n1], n1, mBase, tid,
                          xhl[buf ^ 1][0], xhl[buf ^ 1][1],
                          whl[buf ^ 1][0], whl[buf ^ 1][1]);
        }

        const unsigned short (*xh)[PITCH] = xhl[buf][0];
        const unsigned short (*xl)[PITCH] = xhl[buf][1];
        const unsigned short (*wh)[PITCH] = whl[buf][0];
        const unsigned short (*wl)[PITCH] = whl[buf][1];

        // A 16x32 bf16 frag: lane-half h: v0-7 K=16h..16h+15
        Frag ahi, alo;
        ahi.q[0] = *(const v4u*)&xh[m0 + l15][16 * half];
        ahi.q[1] = *(const v4u*)&xh[m0 + l15][16 * half + 8];
        alo.q[0] = *(const v4u*)&xl[m0 + l15][16 * half];
        alo.q[1] = *(const v4u*)&xl[m0 + l15][16 * half + 8];

        // B 32x16 bf16 frag: lanes 0-15 K=0-15, lanes 16-31 K=16-31; N = row of wT
        Frag bh0, bl0, bh1, bl1;
        {
            const v4u* pb = (const v4u*)&wh[l15][16 * half];
            bh0.q[0] = pb[0]; bh0.q[1] = pb[1];
            const v4u* qb = (const v4u*)&wh[l15 + 16][16 * half];
            bh1.q[0] = qb[0]; bh1.q[1] = qb[1];
            const v4u* pl = (const v4u*)&wl[l15][16 * half];
            bl0.q[0] = pl[0]; bl0.q[1] = pl[1];
            const v4u* ql = (const v4u*)&wl[l15 + 16][16 * half];
            bl1.q[0] = ql[0]; bl1.q[1] = ql[1];
        }

        acc0 = __builtin_amdgcn_wmma_f32_16x16x32_bf16(false, ahi.v, false, bh0.v, (short)0, acc0, false, false);
        acc0 = __builtin_amdgcn_wmma_f32_16x16x32_bf16(false, alo.v, false, bh0.v, (short)0, acc0, false, false);
        acc0 = __builtin_amdgcn_wmma_f32_16x16x32_bf16(false, ahi.v, false, bl0.v, (short)0, acc0, false, false);
        acc1 = __builtin_amdgcn_wmma_f32_16x16x32_bf16(false, ahi.v, false, bh1.v, (short)0, acc1, false, false);
        acc1 = __builtin_amdgcn_wmma_f32_16x16x32_bf16(false, alo.v, false, bh1.v, (short)0, acc1, false, false);
        acc1 = __builtin_amdgcn_wmma_f32_16x16x32_bf16(false, ahi.v, false, bl1.v, (short)0, acc1, false, false);
    }

    // Store [16 x 32] slab. D: VGPR v -> row v (lanes 0-15) / v+8 (lanes 16-31).
    {
        float* dst = out + (mBase + m0) * NDIM + (size_t)kblk * BS;
        #pragma unroll
        for (int v = 0; v < 8; ++v) {
            const int row = v + 8 * half;
            dst[(size_t)row * NDIM + l15]      = acc0[v];
            dst[(size_t)row * NDIM + 16 + l15] = acc1[v];
        }
    }
}

// ---------------------------------------------------------------------------
extern "C" void kernel_launch(void* const* d_in, const int* in_sizes, int n_in,
                              void* d_out, int out_size, void* d_ws, size_t ws_size,
                              hipStream_t stream) {
    const float* x    = (const float*)d_in[0];   // [M, 4096] f32
    const float* wblk = (const float*)d_in[1];   // [nnz, 32, 32] f32
    const int*   cs   = (const int*)d_in[2];     // [nnz] input-block idx
    const int*   ks   = (const int*)d_in[3];     // [nnz] output-block idx

    const long long xElems = (long long)in_sizes[0];
    const long long wElems = (long long)in_sizes[1];
    const int nnz = in_sizes[2];
    const int M   = (int)(xElems / KDIM);

    char* base = (char*)d_ws;
    int* col_ptr = (int*)base;                               // 129 ints
    int* col_idx = (int*)(base + 1024);                      // nnz ints
    unsigned short* xhi = (unsigned short*)(base + 65536);   // M*K bf16
    unsigned short* xlo = xhi + xElems;
    unsigned short* whT = xlo + xElems;                      // nnz*32*32 bf16 (transposed blocks)
    unsigned short* wlT = whT + wElems;

    const long long n4 = xElems / 4;
    SparseLinear_split_x<<<(unsigned)((n4 + 255) / 256), 256, 0, stream>>>(x, xhi, xlo, n4);
    SparseLinear_split_wT<<<nnz, 256, 0, stream>>>(wblk, whT, wlT);
    SparseLinear_build_csc<<<1, 128, 0, stream>>>(ks, nnz, col_ptr, col_idx);

    dim3 grid(OB, M / MTILE);
    SparseLinear_bsmm<<<grid, NTHREADS, 0, stream>>>(
        xhi, xlo, whT, wlT, cs, col_ptr, col_idx, (float*)d_out);
}
